// MDCT_43258910605323
// MI455X (gfx1250) — matrix operand order repaired and checked by reading, not compiled, not run on hardware
//
#include <hip/hip_runtime.h>

typedef __attribute__((ext_vector_type(4)))  float        v4f;
typedef __attribute__((ext_vector_type(8)))  float        v8f;
typedef __attribute__((ext_vector_type(8)))  _Float16     v8h;
typedef __attribute__((ext_vector_type(16))) _Float16     v16h;
typedef __attribute__((ext_vector_type(4)))  unsigned int v4u;
typedef __attribute__((ext_vector_type(8)))  int          v8i;
typedef __attribute__((ext_vector_type(4)))  int          v4i;

#if defined(__has_builtin)
#if __has_builtin(__builtin_amdgcn_tensor_load_to_lds) && __has_builtin(__builtin_amdgcn_s_wait_tensorcnt)
#define USE_TDM 1
#endif
#endif
#ifndef USE_TDM
#define USE_TDM 0
#endif

namespace {
constexpr int      kF       = 2048;            // MDCT frame length (GEMM K)
constexpr int      kN       = 1024;            // output bins (GEMM N)
constexpr unsigned kT       = 1u << 20;        // samples per batch
constexpr int      kB       = 8;               // batches
constexpr int      kFrames  = 1025;            // frames per batch
constexpr int      kRows    = kB * kFrames;    // 8200 GEMM rows (M)
constexpr double   kPI      = 3.14159265358979323846;
constexpr float    kScale   = 0.04419417382415922f; // sqrt(2/1024)
constexpr int      kKB      = kF / 32;         // 64 K-blocks of 32
constexpr int      kBlkM    = 128;             // rows per workgroup
constexpr int      kBlkN    = 256;             // cols per workgroup
constexpr int      kGridM   = (kRows + kBlkM - 1) / kBlkM;  // 65
constexpr int      kGridN   = kN / kBlkN;                   // 4
}

// ---------------------------------------------------------------------------
// Phase 1: basis matrix split into f16 hi/lo, stored in WMMA B-fragment order.
//   Bm[n][k] = window[n]*sqrt(2/N)*cos(pi*((2k+1)(2n+1+N) mod 8N)/(4N))
// Packed layout: flat = ((kb*1024 + k)*2 + khalf)*16 + j, n = kb*32+khalf*16+j.
// => each lane's fragment is 16 contiguous f16, and any (kb, col-range) panel
//    is one contiguous chunk: ideal for a TDM tensor_load_to_lds copy.
// ---------------------------------------------------------------------------
__global__ void mdct_basis_kernel(const float* __restrict__ win,
                                  _Float16* __restrict__ Bh,
                                  _Float16* __restrict__ Bl) {
  int idx = blockIdx.x * blockDim.x + threadIdx.x;
  if (idx >= kF * kN) return;
  int n = idx >> 10;                 // K index 0..2047
  int k = idx & (kN - 1);            // col   0..1023
  int p = ((2 * k + 1) * (2 * n + 1 + kN)) & (8 * kN - 1);  // exact mod 8192
  float val = win[n] * kScale * (float)cos((double)p * (kPI / 4096.0));

  _Float16 h = (_Float16)val;
  _Float16 l = (_Float16)(val - (float)h);

  int kb    = n >> 5;
  int khalf = (n >> 4) & 1;
  int j     = n & 15;
  size_t flat = ((size_t)(kb * kN + k) * 2 + khalf) * 16 + j;
  Bh[flat] = h;
  Bl[flat] = l;
}

// ---------------------------------------------------------------------------
// Phase 2: LDS-tiled split-f16 WMMA GEMM with TDM-staged B panels.
//   out(8200x1024) = A(8200x2048) * B,  acc += Ah*Bh + Ah*Bl + Al*Bh (f32 acc)
// Workgroup: 512 threads = 16 waves (4 row-waves x 4 col-waves) -> 128x256 tile.
// ---------------------------------------------------------------------------
struct ARow {
  const float* xb;   // x + b*T
  int  fOff;         // x index of frame element n=0 (may be negative)
  bool valid;
};

__device__ __forceinline__ ARow makeRow(const float* x, int row) {
  ARow r;
  int b = row / kFrames;
  if (b > kB - 1) b = kB - 1;            // clamp OOB tail rows
  int f   = row - b * kFrames;
  r.valid = row < kRows;
  r.fOff  = f * 1024 - 1024;
  r.xb    = x + (size_t)b * kT;
  return r;
}

// Load 8 consecutive frame samples (8-aligned => validity uniform per group,
// never straddles zero-pad boundaries). Branchless clamp + cndmask.
__device__ __forceinline__ void loadG(const ARow& r, int g, float* d) {
  int idx = r.fOff + g;
  bool v  = r.valid && ((unsigned)idx < kT - 7u);
  unsigned c = v ? (unsigned)idx : 0u;
  v4f a = *(const v4f*)(r.xb + c);
  v4f b = *(const v4f*)(r.xb + c + 4);
  d[0] = v ? a.x : 0.0f;  d[1] = v ? a.y : 0.0f;
  d[2] = v ? a.z : 0.0f;  d[3] = v ? a.w : 0.0f;
  d[4] = v ? b.x : 0.0f;  d[5] = v ? b.y : 0.0f;
  d[6] = v ? b.z : 0.0f;  d[7] = v ? b.w : 0.0f;
}

#if USE_TDM
// One 2D TDM descriptor copies hi(16KB) + lo(16KB) B panels: tile 2048x2 of 8B
// elements, row stride = (Bl - Bh) = 4MB = 524288 * 8B. (D# per ISA 8.3/8.4.)
__device__ __forceinline__ void tdm_load_bpanel(const _Float16* gsrc, void* ldst) {
  unsigned long long ga = (unsigned long long)(size_t)gsrc;
  v4u g0;
  g0[0] = 1u;                                            // count=1 (valid D#)
  g0[1] = (unsigned)(size_t)ldst;                        // lds_addr
  g0[2] = (unsigned)(ga & 0xffffffffu);                  // global_addr[31:0]
  g0[3] = (unsigned)((ga >> 32) & 0x1ffffffu) | (2u << 30); // addr[56:32], type=2
  v8i g1;
  g1[0] = 0x30000;               // workgroup_mask=0, data_size=3 (8 bytes)
  g1[1] = (int)(2048u << 16);    // tensor_dim0[15:0]=2048 (bits 63:48)
  g1[2] = (int)(2u << 16);       // tensor_dim0[31:16]=0, tensor_dim1=2
  g1[3] = (int)(2048u << 16);    // tensor_dim1 hi=0, tile_dim0=2048
  g1[4] = 2;                     // tile_dim1=2, tile_dim2=0
  g1[5] = 524288;                // tensor_dim0_stride[31:0] (8B units)
  g1[6] = 0;                     // stride hi, tensor_dim1_stride lo
  g1[7] = 0;
  v4i z4 = {0, 0, 0, 0};         // groups 2/3 unused (<=2D tensor)
  v8i z8 = {0, 0, 0, 0, 0, 0, 0, 0};
  __builtin_amdgcn_tensor_load_to_lds(g0, g1, z4, z4, z8, 0);
}
#endif

#define MM(acc, Af, Bf) \
  acc = __builtin_amdgcn_wmma_f32_16x16x32_f16(false, Af, false, Bf, (short)0, acc, false, false)

__global__ __launch_bounds__(512)
void mdct_wmma_kernel(const float* __restrict__ x,
                      const _Float16* __restrict__ Bh,
                      const _Float16* __restrict__ Bl,
                      float* __restrict__ out) {
  // LDS: A tile 128x32 hi/lo fragments (8 KB each), B panel [hi|lo] 32 KB.
  // Double buffered: 96 KB total.
  __shared__ alignas(64) _Float16 AH[2][kBlkM * 32];
  __shared__ alignas(64) _Float16 AL[2][kBlkM * 32];
  __shared__ alignas(64) _Float16 Bp[2][2][kBlkN * 32];   // [buf][hi/lo][8192]

  const int tid     = threadIdx.x;
  const int lane    = tid & 31;
  const int wid     = tid >> 5;        // 0..15
  const int waveRow = wid >> 2;        // 0..3 (32 rows each)
  const int waveCol = wid & 3;         // 0..3 (64 cols each)
  const int laneM   = lane & 15;
  const int kh      = lane >> 4;       // K-half select

  const int rBase = blockIdx.x * kBlkM;
  const int cBase = blockIdx.y * kBlkN;

  // ---- A staging roles: thread -> (row, K-octet)
  const int  sRow  = tid >> 2;         // 0..127
  const int  sOct  = tid & 3;          // 0..3 : K = 8*sOct .. +7
  const int  sKh   = sOct >> 1;        // fragment K-half
  const int  sHalf = sOct & 1;         // element half (0->e0..7, 1->e8..15)
  const int  sFlat = (sRow * 2 + sKh) * 16 + sHalf * 8;
  const ARow ar    = makeRow(x, rBase + sRow);

  const _Float16* bpanel0 = Bh + (size_t)cBase * 32;   // hi chunk; lo = +4MB

#if !USE_TDM
  const _Float16* bhg = Bh + (size_t)cBase * 32 + (size_t)tid * 16;
  const _Float16* blg = Bl + (size_t)cBase * 32 + (size_t)tid * 16;
#endif

  auto stageB = [&](int kb, int buf) {
#if USE_TDM
    if (wid == 0)
      tdm_load_bpanel(bpanel0 + (size_t)kb * kN * 32, &Bp[buf][0][0]);
#else
    const size_t go = (size_t)kb * kN * 32;
    *(v16h*)&Bp[buf][0][tid * 16] = *(const v16h*)(bhg + go);
    *(v16h*)&Bp[buf][1][tid * 16] = *(const v16h*)(blg + go);
#endif
  };
  auto stageA = [&](int kb, int buf) {
    float d[8];
    loadG(ar, kb * 32 + sOct * 8, d);
    v8h h, l;
#pragma unroll
    for (int i = 0; i < 8; ++i) {
      _Float16 hi = (_Float16)d[i];
      h[i] = hi;
      l[i] = (_Float16)(d[i] - (float)hi);
    }
    *(v8h*)&AH[buf][sFlat] = h;
    *(v8h*)&AL[buf][sFlat] = l;
  };

  stageB(0, 0);
  stageA(0, 0);
#if USE_TDM
  if (wid == 0) __builtin_amdgcn_s_wait_tensorcnt(0);
#endif
  __syncthreads();

  v8f acc[2][4] = {};

  for (int kb = 0; kb < kKB; ++kb) {
    const int cur = kb & 1;

    // A fragments for this wave's two 16-row groups
    const int a0 = ((waveRow * 32 + laneM) * 2 + kh) * 16;
    const int a1 = ((waveRow * 32 + 16 + laneM) * 2 + kh) * 16;
    v16h ah0 = *(const v16h*)&AH[cur][a0];
    v16h al0 = *(const v16h*)&AL[cur][a0];
    v16h ah1 = *(const v16h*)&AH[cur][a1];
    v16h al1 = *(const v16h*)&AL[cur][a1];

    // B fragments: 4 col tiles of 16
    v16h bhf[4], blf[4];
#pragma unroll
    for (int t = 0; t < 4; ++t) {
      const int cc = waveCol * 64 + 16 * t + laneM;
      const int bo = (cc * 2 + kh) * 16;
      bhf[t] = *(const v16h*)&Bp[cur][0][bo];
      blf[t] = *(const v16h*)&Bp[cur][1][bo];
    }

    // prefetch next K-step into the other buffer (hidden under WMMAs)
    if (kb + 1 < kKB) {
      stageB(kb + 1, cur ^ 1);
      stageA(kb + 1, cur ^ 1);
    }

#pragma unroll
    for (int t = 0; t < 4; ++t) {
      MM(acc[0][t], ah0, bhf[t]); MM(acc[0][t], ah0, blf[t]); MM(acc[0][t], al0, bhf[t]);
      MM(acc[1][t], ah1, bhf[t]); MM(acc[1][t], ah1, blf[t]); MM(acc[1][t], al1, bhf[t]);
    }

#if USE_TDM
    if (wid == 0) __builtin_amdgcn_s_wait_tensorcnt(0);
#endif
    __syncthreads();
  }

  // D layout: VGPR j -> row rf*16 + j + 8*kh; col 16*t + laneM (within wave tile)
#pragma unroll
  for (int rf = 0; rf < 2; ++rf) {
#pragma unroll
    for (int j = 0; j < 8; ++j) {
      const int r = rBase + waveRow * 32 + rf * 16 + j + 8 * kh;
      if (r < kRows) {
        float* o = out + (size_t)r * kN + (cBase + waveCol * 64 + laneM);
        o[0]  = acc[rf][0][j];
        o[16] = acc[rf][1][j];
        o[32] = acc[rf][2][j];
        o[48] = acc[rf][3][j];
      }
    }
  }
}

// ---------------------------------------------------------------------------
extern "C" void kernel_launch(void* const* d_in, const int* in_sizes, int n_in,
                              void* d_out, int out_size, void* d_ws, size_t ws_size,
                              hipStream_t stream) {
  const float* x   = (const float*)d_in[0];   // (8, 1<<20) float32
  const float* win = (const float*)d_in[1];   // (2048,)    float32
  _Float16* Bh = (_Float16*)d_ws;             // 2048*1024 f16 = 4 MB
  _Float16* Bl = Bh + (size_t)kF * kN;        // 4 MB, exactly 4MB after Bh
  float* out = (float*)d_out;                 // (8, 1025, 1024) float32

  mdct_basis_kernel<<<(kF * kN + 255) / 256, 256, 0, stream>>>(win, Bh, Bl);

  dim3 grid(kGridM, kGridN);                  // 65 x 4 workgroups
  mdct_wmma_kernel<<<grid, 512, 0, stream>>>(x, Bh, Bl, out);
}
